// StructureModule_44375602102630
// MI455X (gfx1250) — compile-verified, matrix operand order checked by reading, not run on hardware
//
#include <hip/hip_runtime.h>

// ---------------------------------------------------------------- types ----
typedef __attribute__((ext_vector_type(16))) __bf16 v16bf;
typedef __attribute__((ext_vector_type(8)))  float  v8f;

union Frag16 { v16bf v; unsigned short u[16]; uint4 q[2]; };

__device__ __forceinline__ unsigned short f2bf(float f) {
    unsigned int u = __float_as_uint(f);
    return (unsigned short)((u + 0x7FFFu + ((u >> 16) & 1u)) >> 16);
}

__device__ __forceinline__ v8f wmma_bf16(v16bf a, v16bf b, v8f c) {
    return __builtin_amdgcn_wmma_f32_16x16x32_bf16(false, a, false, b, (short)0, c, false, false);
}

// A/B fragment from row-major bf16 matrix (row = 32*2-byte aligned pointer).
// lane l: m(or n) = l&15, half = l>>4 ; element e maps to K = 16*(e>>3)+8*half+(e&7)
__device__ __forceinline__ v16bf frag_row(const unsigned short* row, int k0, int half) {
    Frag16 f;
    f.q[0] = *(const uint4*)(row + k0 + 8 * half);
    f.q[1] = *(const uint4*)(row + k0 + 16 + 8 * half);
    return f.v;
}
__device__ __forceinline__ v16bf frag_sw(const unsigned short* p) {
    Frag16 f;
    const uint4* q = (const uint4*)p;
    f.q[0] = q[0]; f.q[1] = q[1];
    return f.v;
}

__device__ __forceinline__ void quat2rot(const float* q, float* R) {
    float w = q[0], x = q[1], y = q[2], z = q[3];
    R[0] = 1.f - 2.f * (y * y + z * z); R[1] = 2.f * (x * y - w * z); R[2] = 2.f * (x * z + w * y);
    R[3] = 2.f * (x * y + w * z); R[4] = 1.f - 2.f * (x * x + z * z); R[5] = 2.f * (y * z - w * x);
    R[6] = 2.f * (x * z - w * y); R[7] = 2.f * (y * z + w * x); R[8] = 1.f - 2.f * (x * x + y * y);
}

// ------------------------------------------------------------ constants ----
#define L 768
#define CS 384
#define CZ 256
#define HN 12
#define CH 16
#define DOUT 3648

// ------------------------------------------------------ weight swizzler ----
// W is K x N row-major f32 -> fragments [kt][ntOff+nt][lane][16] bf16 (zero pad)
__global__ void k_swizzle(const float* __restrict__ W, int K, int N,
                          unsigned short* __restrict__ out,
                          int KT, int NTloc, int NTtot, int ntOff) {
    long total = (long)KT * NTloc * 512;
    for (long x = (long)blockIdx.x * blockDim.x + threadIdx.x; x < total;
         x += (long)gridDim.x * blockDim.x) {
        int e = (int)(x & 15);
        int lane = (int)((x >> 4) & 31);
        long tile = x >> 9;
        int nt = (int)(tile % NTloc);
        int kt = (int)(tile / NTloc);
        int half = lane >> 4;
        int n = nt * 16 + (lane & 15);
        int k = kt * 32 + 16 * (e >> 3) + 8 * half + (e & 7);
        float v = (k < K && n < N) ? W[(long)k * N + n] : 0.f;
        out[((long)(kt * NTtot + ntOff + nt) * 32 + lane) * 16 + e] = f2bf(v);
    }
}

// ------------------------------------------------------------ GEMM (ws) ----
// C[M=768 x N] = A(bf16, lda) @ Wsw + biasv + addend ; optional relu
// Dual accumulators break the WMMA->WMMA D-dependency chain (avoids the
// 1+4 co-exec hazard NOPs per issue, ISA 7.12.1).
__global__ void k_gemm_ws(const unsigned short* __restrict__ A, int lda,
                          const unsigned short* __restrict__ Wsw,
                          const float* __restrict__ biasv,
                          const float* __restrict__ addend, int ldadd,
                          float* __restrict__ C, int ldc,
                          int N, int KT, int relu) {
    int l = threadIdx.x;
    int NT = gridDim.x;
    int n0 = blockIdx.x * 16, m0 = blockIdx.y * 16;
    int half = l >> 4, lm = l & 15;
    v8f acc0, acc1;
    for (int i = 0; i < 8; ++i) { acc0[i] = 0.f; acc1[i] = 0.f; }
    const unsigned short* arow = A + (long)(m0 + lm) * lda;
    int kt = 0;
    for (; kt + 1 < KT; kt += 2) {
        v16bf a0 = frag_row(arow, kt * 32, half);
        v16bf b0 = frag_sw(Wsw + ((long)(kt * NT + blockIdx.x) * 32 + l) * 16);
        v16bf a1 = frag_row(arow, (kt + 1) * 32, half);
        v16bf b1 = frag_sw(Wsw + ((long)((kt + 1) * NT + blockIdx.x) * 32 + l) * 16);
        acc0 = wmma_bf16(a0, b0, acc0);
        acc1 = wmma_bf16(a1, b1, acc1);
    }
    if (kt < KT) {
        v16bf a0 = frag_row(arow, kt * 32, half);
        v16bf b0 = frag_sw(Wsw + ((long)(kt * NT + blockIdx.x) * 32 + l) * 16);
        acc0 = wmma_bf16(a0, b0, acc0);
    }
    int n = n0 + lm;
    if (n >= N) return;
    float bv = biasv ? biasv[n] : 0.f;
    for (int v = 0; v < 8; ++v) {
        int m = m0 + v + 8 * half;
        float r = acc0[v] + acc1[v] + bv;
        if (addend) r += addend[(long)m * ldadd + n];
        if (relu) r = fmaxf(r, 0.f);
        C[(long)m * ldc + n] = r;
    }
}

// ------------------------------------------------- attention logits GEMM ----
// per head h: logits[h,i,j] = Aatt[h,i,:32] . Batt[h,j,:32] + biasS + rowt + colt
__global__ void k_gemm_att(const unsigned short* __restrict__ Aatt,
                           const unsigned short* __restrict__ Batt,
                           const float* __restrict__ biasS,
                           const float* __restrict__ rowt,
                           const float* __restrict__ colt,
                           float* __restrict__ logits) {
    int l = threadIdx.x;
    int h = blockIdx.z;
    int m0 = blockIdx.y * 16, n0 = blockIdx.x * 16;
    int half = l >> 4, lm = l & 15;
    v16bf a = frag_row(Aatt + ((long)h * L + m0 + lm) * 32, 0, half);
    v16bf b = frag_row(Batt + ((long)h * L + n0 + lm) * 32, 0, half);
    v8f acc; for (int i = 0; i < 8; ++i) acc[i] = 0.f;
    acc = wmma_bf16(a, b, acc);
    int n = n0 + lm;
    float ct = colt[h * L + n];
    long base = (long)h * L * L;
    for (int v = 0; v < 8; ++v) {
        int m = m0 + v + 8 * half;
        logits[base + (long)m * L + n] =
            acc[v] + biasS[base + (long)m * L + n] + rowt[h * L + m] + ct;
    }
}

// --------------------------------------------------------- GEMM (A x Bt) ----
// B stored N x K row-major bf16 (b128 loads).  Batched over blockIdx.z.
// Used for opair: B = z[i]^T streams ~300MB/layer from HBM -> prefetch ahead.
__global__ void k_gemm_nt(const unsigned short* __restrict__ A, int lda, long sA,
                          const unsigned short* __restrict__ B, int ldb, long sB,
                          float* __restrict__ C, int ldc, long sC,
                          int Mstore, int KT) {
    int l = threadIdx.x;
    int z = blockIdx.z;
    int m0 = blockIdx.y * 16, n0 = blockIdx.x * 16;
    int half = l >> 4, lm = l & 15;
    const unsigned short* arow = A + z * sA + (long)(m0 + lm) * lda;
    const unsigned short* brow = B + z * sB + (long)(n0 + lm) * ldb;
    v8f acc0, acc1;
    for (int i = 0; i < 8; ++i) { acc0[i] = 0.f; acc1[i] = 0.f; }
    int kt = 0;
    for (; kt + 1 < KT; kt += 2) {
        if (kt + 3 < KT) {
            __builtin_prefetch(brow + (kt + 2) * 32, 0, 1);
            __builtin_prefetch(brow + (kt + 3) * 32, 0, 1);
        }
        v16bf a0 = frag_row(arow, kt * 32, half);
        v16bf b0 = frag_row(brow, kt * 32, half);
        v16bf a1 = frag_row(arow, (kt + 1) * 32, half);
        v16bf b1 = frag_row(brow, (kt + 1) * 32, half);
        acc0 = wmma_bf16(a0, b0, acc0);
        acc1 = wmma_bf16(a1, b1, acc1);
    }
    if (kt < KT) {
        v16bf a0 = frag_row(arow, kt * 32, half);
        v16bf b0 = frag_row(brow, kt * 32, half);
        acc0 = wmma_bf16(a0, b0, acc0);
    }
    int n = n0 + lm;
    float* Cz = C + z * sC;
    for (int v = 0; v < 8; ++v) {
        int m = m0 + v + 8 * half;
        if (m < Mstore) Cz[(long)m * ldc + n] = acc0[v] + acc1[v];
    }
}

// --------------------------------------------------------- GEMM (A x B) ----
// B f32, K x N with row stride ldb (narrow N: o / opg).  Batched over z.
__global__ void k_gemm_nn(const unsigned short* __restrict__ A, int lda, long sA,
                          const float* __restrict__ B, int ldb, long sB,
                          float* __restrict__ C, int ldc, long sC,
                          int N, int KT) {
    int l = threadIdx.x;
    int z = blockIdx.z;
    int m0 = blockIdx.y * 16, n0 = blockIdx.x * 16;
    int half = l >> 4, lm = l & 15;
    int n = n0 + lm;
    const unsigned short* arow = A + z * sA + (long)(m0 + lm) * lda;
    const float* Bz = B + z * sB;
    v8f acc0, acc1;
    for (int i = 0; i < 8; ++i) { acc0[i] = 0.f; acc1[i] = 0.f; }
    int kt = 0;
    for (; kt + 1 < KT; kt += 2) {
        v16bf a0 = frag_row(arow, kt * 32, half);
        v16bf a1 = frag_row(arow, (kt + 1) * 32, half);
        Frag16 fb0, fb1;
        for (int e = 0; e < 16; ++e) {
            int kk = 16 * (e >> 3) + 8 * half + (e & 7);
            fb0.u[e] = (n < N) ? f2bf(Bz[(long)(kt * 32 + kk) * ldb + n]) : (unsigned short)0;
            fb1.u[e] = (n < N) ? f2bf(Bz[(long)((kt + 1) * 32 + kk) * ldb + n]) : (unsigned short)0;
        }
        acc0 = wmma_bf16(a0, fb0.v, acc0);
        acc1 = wmma_bf16(a1, fb1.v, acc1);
    }
    if (kt < KT) {
        v16bf a0 = frag_row(arow, kt * 32, half);
        Frag16 fb0;
        for (int e = 0; e < 16; ++e) {
            int kk = 16 * (e >> 3) + 8 * half + (e & 7);
            fb0.u[e] = (n < N) ? f2bf(Bz[(long)(kt * 32 + kk) * ldb + n]) : (unsigned short)0;
        }
        acc0 = wmma_bf16(a0, fb0.v, acc0);
    }
    if (n >= N) return;
    float* Cz = C + z * sC;
    for (int v = 0; v < 8; ++v) {
        int m = m0 + v + 8 * half;
        Cz[(long)m * ldc + n] = acc0[v] + acc1[v];
    }
}

// ----------------------------------------------------- prep: single rep ----
__global__ void k_prep_single(const float* __restrict__ sfea, const float* __restrict__ encd,
                              const float* __restrict__ g, const float* __restrict__ b,
                              float* __restrict__ s_init, unsigned short* __restrict__ s_init_b,
                              unsigned short* __restrict__ rc) {
    int i = blockIdx.x, t = threadIdx.x; // 128 threads
    const float* row = sfea + (long)i * CS;
    float x0 = row[t], x1 = row[t + 128], x2 = row[t + 256];
    __shared__ float red[128];
    red[t] = x0 + x1 + x2; __syncthreads();
    for (int o = 64; o > 0; o >>= 1) { if (t < o) red[t] += red[t + o]; __syncthreads(); }
    float mean = red[0] / (float)CS; __syncthreads();
    float d0 = x0 - mean, d1 = x1 - mean, d2 = x2 - mean;
    red[t] = d0 * d0 + d1 * d1 + d2 * d2; __syncthreads();
    for (int o = 64; o > 0; o >>= 1) { if (t < o) red[t] += red[t + o]; __syncthreads(); }
    float rstd = rsqrtf(red[0] / (float)CS + 1e-5f);
    int c;
    float v;
    c = t;        v = d0 * rstd * g[c] + b[c];
    s_init[(long)i * CS + c] = v; s_init_b[(long)i * CS + c] = f2bf(v);
    rc[(long)i * 416 + c] = f2bf(fmaxf(v, 0.f));
    c = t + 128;  v = d1 * rstd * g[c] + b[c];
    s_init[(long)i * CS + c] = v; s_init_b[(long)i * CS + c] = f2bf(v);
    rc[(long)i * 416 + c] = f2bf(fmaxf(v, 0.f));
    c = t + 256;  v = d2 * rstd * g[c] + b[c];
    s_init[(long)i * CS + c] = v; s_init_b[(long)i * CS + c] = f2bf(v);
    rc[(long)i * 416 + c] = f2bf(fmaxf(v, 0.f));
    if (t < 32) rc[(long)i * 416 + 384 + t] = f2bf(fmaxf(encd[(long)i * 32 + t], 0.f));
}

// --------------------------------------------- prep: pair LN + transpose ----
// zbt[i][d][j] = bf16(LN(pfea[i][j][:])[d]) ; biasS[h][i][j] = (z . Wb[:,h])/sqrt3
__global__ void k_prep_pair(const float* __restrict__ pfea,
                            const float* __restrict__ g, const float* __restrict__ b,
                            const float* __restrict__ Wb,
                            unsigned short* __restrict__ zbt, float* __restrict__ biasS) {
    int i = blockIdx.y;
    int j0 = blockIdx.x * 16;
    int t = threadIdx.x; // 256
    int jl = t >> 4, dl = t & 15;
    __shared__ float red[256];
    __shared__ float mean[16], rstd[16];
    const float* base = pfea + ((long)i * L + j0) * CZ;
    float s = 0.f, s2 = 0.f;
    for (int d = dl; d < CZ; d += 16) {
        float x = base[(long)jl * CZ + d];
        s += x; s2 += x * x;
    }
    red[t] = s; __syncthreads();
    for (int o = 8; o > 0; o >>= 1) { if (dl < o) red[t] += red[t + o]; __syncthreads(); }
    if (dl == 0) mean[jl] = red[jl * 16] / (float)CZ;
    __syncthreads();
    red[t] = s2; __syncthreads();
    for (int o = 8; o > 0; o >>= 1) { if (dl < o) red[t] += red[t + o]; __syncthreads(); }
    if (dl == 0) {
        float m = mean[jl];
        rstd[jl] = rsqrtf(red[jl * 16] / (float)CZ - m * m + 1e-5f);
    }
    __syncthreads();
    // pass B: j = j0 + (t&15), d-group = t>>4
    int jw = t & 15, dg = t >> 4;
    const float* row = pfea + ((long)i * L + j0 + jw) * CZ;
    float m = mean[jw], r = rstd[jw];
    float acc[HN];
    for (int h = 0; h < HN; ++h) acc[h] = 0.f;
    for (int db = 0; db < 16; ++db) {
        int d = db * 16 + dg;
        float z = (row[d] - m) * r * g[d] + b[d];
        zbt[((long)i * CZ + d) * L + j0 + jw] = f2bf(z);
        for (int h = 0; h < HN; ++h) acc[h] += z * Wb[d * HN + h];
    }
    const float inv_s3 = 0.57735026919f;
    for (int h = 0; h < HN; ++h) {
        __syncthreads();
        red[t] = acc[h]; __syncthreads();
        for (int o = 128; o >= 16; o >>= 1) { if (t < o) red[t] += red[t + o]; __syncthreads(); }
        if (t < 16) biasS[((long)h * L + i) * L + j0 + t] = red[t] * inv_s3;
    }
}

// ---------------------------------------------------------------- misc -----
__global__ void k_cvt(const float* __restrict__ in, unsigned short* __restrict__ out,
                      long n, int relu) {
    for (long x = (long)blockIdx.x * blockDim.x + threadIdx.x; x < n;
         x += (long)gridDim.x * blockDim.x) {
        float v = in[x];
        if (relu) v = fmaxf(v, 0.f);
        out[x] = f2bf(v);
    }
}

__global__ void k_zeropad_abf(unsigned short* __restrict__ abf) {
    long n = (long)L * 4 * L;
    for (long x = (long)blockIdx.x * blockDim.x + threadIdx.x; x < n;
         x += (long)gridDim.x * blockDim.x) {
        long i = x / (4 * L);
        long r = x % (4 * L);
        abf[i * (16 * L) + 12 * L + r] = 0;
    }
}

__global__ void k_init_qt(float* __restrict__ quat, float* __restrict__ trsl) {
    int i = blockIdx.x * blockDim.x + threadIdx.x;
    if (i < L) {
        quat[i * 4 + 0] = 1.f; quat[i * 4 + 1] = 0.f;
        quat[i * 4 + 2] = 0.f; quat[i * 4 + 3] = 0.f;
        trsl[i * 3 + 0] = 0.f; trsl[i * 3 + 1] = 0.f; trsl[i * 3 + 2] = 0.f;
    }
}

// residual + layernorm: s_out = LN(s_old + delta)*g + b ; also bf16 copy
__global__ void k_ln_res(const float* __restrict__ sold, const float* __restrict__ delta,
                         const float* __restrict__ g, const float* __restrict__ bb,
                         float* __restrict__ sout, unsigned short* __restrict__ sbout) {
    int i = blockIdx.x, t = threadIdx.x; // 128
    const float* a = sold + (long)i * CS;
    const float* d = delta + (long)i * CS;
    float x0 = a[t] + d[t], x1 = a[t + 128] + d[t + 128], x2 = a[t + 256] + d[t + 256];
    __shared__ float red[128];
    red[t] = x0 + x1 + x2; __syncthreads();
    for (int o = 64; o > 0; o >>= 1) { if (t < o) red[t] += red[t + o]; __syncthreads(); }
    float mean = red[0] / (float)CS; __syncthreads();
    float d0 = x0 - mean, d1 = x1 - mean, d2 = x2 - mean;
    red[t] = d0 * d0 + d1 * d1 + d2 * d2; __syncthreads();
    for (int o = 64; o > 0; o >>= 1) { if (t < o) red[t] += red[t + o]; __syncthreads(); }
    float rstd = rsqrtf(red[0] / (float)CS + 1e-5f);
    int c; float v;
    c = t;       v = d0 * rstd * g[c] + bb[c]; sout[(long)i * CS + c] = v; sbout[(long)i * CS + c] = f2bf(v);
    c = t + 128; v = d1 * rstd * g[c] + bb[c]; sout[(long)i * CS + c] = v; sbout[(long)i * CS + c] = f2bf(v);
    c = t + 256; v = d2 * rstd * g[c] + bb[c]; sout[(long)i * CS + c] = v; sbout[(long)i * CS + c] = f2bf(v);
}

// --------------------------------------------- IPA point transformation ----
// builds packed attention operands, vpg buffer, row/col distance terms
__global__ void k_point(const float* __restrict__ proj, const float* __restrict__ quat,
                        const float* __restrict__ trsl, const float* __restrict__ hw,
                        unsigned short* __restrict__ Aatt, unsigned short* __restrict__ Batt,
                        float* __restrict__ rowt, float* __restrict__ colt,
                        float* __restrict__ vpg) {
    int i = blockIdx.x, t = threadIdx.x; // 64
    __shared__ float R[9], tr[3], gb[HN];
    __shared__ float qpg[HN][12], kpg[HN][12];
    if (t == 0) {
        quat2rot(quat + i * 4, R);
        tr[0] = trsl[i * 3]; tr[1] = trsl[i * 3 + 1]; tr[2] = trsl[i * 3 + 2];
    }
    if (t < HN) {
        // gb = 0.5 * softplus(hw) * wc / sqrt(3)
        gb[t] = 0.5f * log1pf(__expf(hw[t])) * 0.23570226039f * 0.57735026919f;
    }
    __syncthreads();
    const float* prow = proj + (long)i * 1152;
    for (int idx = t; idx < 192; idx += 64) {
        const float* sp;
        float v0, v1, v2, g0, g1, g2;
        if (idx < 48) {
            int h = idx >> 2, p = idx & 3;
            sp = prow + 576 + h * 12 + p * 3;
            v0 = sp[0]; v1 = sp[1]; v2 = sp[2];
            g0 = R[0] * v0 + R[1] * v1 + R[2] * v2 + tr[0];
            g1 = R[3] * v0 + R[4] * v1 + R[5] * v2 + tr[1];
            g2 = R[6] * v0 + R[7] * v1 + R[8] * v2 + tr[2];
            qpg[h][p * 3 + 0] = g0; qpg[h][p * 3 + 1] = g1; qpg[h][p * 3 + 2] = g2;
        } else if (idx < 96) {
            int j = idx - 48; int h = j >> 2, p = j & 3;
            sp = prow + 720 + h * 12 + p * 3;
            v0 = sp[0]; v1 = sp[1]; v2 = sp[2];
            g0 = R[0] * v0 + R[1] * v1 + R[2] * v2 + tr[0];
            g1 = R[3] * v0 + R[4] * v1 + R[5] * v2 + tr[1];
            g2 = R[6] * v0 + R[7] * v1 + R[8] * v2 + tr[2];
            kpg[h][p * 3 + 0] = g0; kpg[h][p * 3 + 1] = g1; kpg[h][p * 3 + 2] = g2;
        } else {
            int j = idx - 96; int h = j >> 3, p = j & 7;
            sp = prow + 864 + h * 24 + p * 3;
            v0 = sp[0]; v1 = sp[1]; v2 = sp[2];
            g0 = R[0] * v0 + R[1] * v1 + R[2] * v2 + tr[0];
            g1 = R[3] * v0 + R[4] * v1 + R[5] * v2 + tr[1];
            g2 = R[6] * v0 + R[7] * v1 + R[8] * v2 + tr[2];
            vpg[(long)i * 288 + h * 24 + p * 3 + 0] = g0;
            vpg[(long)i * 288 + h * 24 + p * 3 + 1] = g1;
            vpg[(long)i * 288 + h * 24 + p * 3 + 2] = g2;
        }
    }
    __syncthreads();
    const float s1 = 0.14433756730f; // 1/(sqrt(16)*sqrt(3))
    for (int idx = t; idx < 192; idx += 64) {
        int h = idx >> 4, c = idx & 15;
        Aatt[((long)h * L + i) * 32 + c] = f2bf(prow[h * 16 + c] * s1);
        Batt[((long)h * L + i) * 32 + c] = f2bf(prow[192 + h * 16 + c]);
    }
    for (int idx = t; idx < 144; idx += 64) {
        int h = idx / 12, e = idx % 12;
        Aatt[((long)h * L + i) * 32 + 16 + e] = f2bf(qpg[h][e] * 2.f * gb[h]);
        Batt[((long)h * L + i) * 32 + 16 + e] = f2bf(kpg[h][e]);
    }
    for (int idx = t; idx < 48; idx += 64) {
        int h = idx >> 2, c = 28 + (idx & 3);
        Aatt[((long)h * L + i) * 32 + c] = 0;
        Batt[((long)h * L + i) * 32 + c] = 0;
    }
    if (t < HN) {
        float sq = 0.f, sk = 0.f;
        for (int e = 0; e < 12; ++e) { sq += qpg[t][e] * qpg[t][e]; sk += kpg[t][e] * kpg[t][e]; }
        rowt[t * L + i] = -gb[t] * sq;
        colt[t * L + i] = -gb[t] * sk;
    }
}

// ------------------------------------------------------------- softmax -----
__global__ void k_softmax(const float* __restrict__ logits, unsigned short* __restrict__ abf) {
    int i = blockIdx.x, h = blockIdx.y, t = threadIdx.x; // 256
    const float* row = logits + ((long)h * L + i) * L;
    unsigned short* orow = abf + ((long)i * 16 + h) * L;
    float v0 = row[t], v1 = row[t + 256], v2 = row[t + 512];
    __shared__ float red[256];
    red[t] = fmaxf(v0, fmaxf(v1, v2)); __syncthreads();
    for (int o = 128; o > 0; o >>= 1) { if (t < o) red[t] = fmaxf(red[t], red[t + o]); __syncthreads(); }
    float mx = red[0]; __syncthreads();
    float e0 = __expf(v0 - mx), e1 = __expf(v1 - mx), e2 = __expf(v2 - mx);
    red[t] = e0 + e1 + e2; __syncthreads();
    for (int o = 128; o > 0; o >>= 1) { if (t < o) red[t] += red[t + o]; __syncthreads(); }
    float inv = 1.f / red[0];
    orow[t] = f2bf(e0 * inv);
    orow[t + 256] = f2bf(e1 * inv);
    orow[t + 512] = f2bf(e2 * inv);
}

// -------------------------------------------------------- opl / onorm ------
__global__ void k_opl(const float* __restrict__ opgin, const float* __restrict__ quat,
                      const float* __restrict__ trsl, float* __restrict__ cat) {
    int i = blockIdx.x, t = threadIdx.x; // 32
    __shared__ float R[9], tr[3];
    if (t == 0) {
        quat2rot(quat + i * 4, R);
        tr[0] = trsl[i * 3]; tr[1] = trsl[i * 3 + 1]; tr[2] = trsl[i * 3 + 2];
    }
    __syncthreads();
    for (int idx = t; idx < 96; idx += 32) {
        int h = idx >> 3, p = idx & 7;
        const float* v = opgin + (long)i * 288 + h * 24 + p * 3;
        float v0 = v[0] - tr[0], v1 = v[1] - tr[1], v2 = v[2] - tr[2];
        float o0 = R[0] * v0 + R[3] * v1 + R[6] * v2; // R^T v
        float o1 = R[1] * v0 + R[4] * v1 + R[7] * v2;
        float o2 = R[2] * v0 + R[5] * v1 + R[8] * v2;
        float* c = cat + (long)i * DOUT;
        c[192 + h * 24 + p * 3 + 0] = o0;
        c[192 + h * 24 + p * 3 + 1] = o1;
        c[192 + h * 24 + p * 3 + 2] = o2;
        c[480 + h * 8 + p] = sqrtf(o0 * o0 + o1 * o1 + o2 * o2 + 1e-8f);
    }
}

// -------------------------------------------------- backbone frame update --
__global__ void k_bb(const float* __restrict__ s, const float* __restrict__ W,
                     const float* __restrict__ bv,
                     float* __restrict__ quat, float* __restrict__ trsl,
                     float* __restrict__ oq, float* __restrict__ ot,
                     float* __restrict__ oqu) {
    int i = blockIdx.x, t = threadIdx.x; // 64
    __shared__ float red[64];
    __shared__ float upd[6];
    for (int o = 0; o < 6; ++o) {
        float acc = 0.f;
        for (int k = t; k < CS; k += 64) acc += s[(long)i * CS + k] * W[k * 6 + o];
        red[t] = acc; __syncthreads();
        for (int off = 32; off > 0; off >>= 1) { if (t < off) red[t] += red[t + off]; __syncthreads(); }
        if (t == 0) upd[o] = red[0] + bv[o];
        __syncthreads();
    }
    if (t == 0) {
        float q0 = quat[i * 4], q1 = quat[i * 4 + 1], q2 = quat[i * 4 + 2], q3 = quat[i * 4 + 3];
        float u0 = upd[0], u1 = upd[1], u2 = upd[2];
        float nq = rsqrtf(1.f + u0 * u0 + u1 * u1 + u2 * u2);
        float w2 = nq, x2 = u0 * nq, y2 = u1 * nq, z2 = u2 * nq;
        float R[9];
        float qold[4] = {q0, q1, q2, q3};
        quat2rot(qold, R);
        float w = q0 * w2 - q1 * x2 - q2 * y2 - q3 * z2;
        float x = q0 * x2 + q1 * w2 + q2 * z2 - q3 * y2;
        float y = q0 * y2 - q1 * z2 + q2 * w2 + q3 * x2;
        float z = q0 * z2 + q1 * y2 - q2 * x2 + q3 * w2;
        float inv = rsqrtf(w * w + x * x + y * y + z * z);
        w *= inv; x *= inv; y *= inv; z *= inv;
        float t0 = trsl[i * 3 + 0] + R[0] * upd[3] + R[1] * upd[4] + R[2] * upd[5];
        float t1 = trsl[i * 3 + 1] + R[3] * upd[3] + R[4] * upd[4] + R[5] * upd[5];
        float t2 = trsl[i * 3 + 2] + R[6] * upd[3] + R[7] * upd[4] + R[8] * upd[5];
        quat[i * 4] = w; quat[i * 4 + 1] = x; quat[i * 4 + 2] = y; quat[i * 4 + 3] = z;
        trsl[i * 3] = t0; trsl[i * 3 + 1] = t1; trsl[i * 3 + 2] = t2;
        oq[i * 4] = w; oq[i * 4 + 1] = x; oq[i * 4 + 2] = y; oq[i * 4 + 3] = z;
        ot[i * 3] = t0; ot[i * 3 + 1] = t1; ot[i * 3 + 2] = t2;
        oqu[i * 4] = w2; oqu[i * 4 + 1] = x2; oqu[i * 4 + 2] = y2; oqu[i * 4 + 3] = z2;
    }
}

// ------------------------------------------------------- final frames ------
__global__ void k_fram(const float* __restrict__ quat, const float* __restrict__ trsl,
                       const float* __restrict__ ang_last, float* __restrict__ outfr) {
    int i = blockIdx.x, t = threadIdx.x; // 32
    __shared__ float R[9];
    if (t == 0) quat2rot(quat + i * 4, R);
    __syncthreads();
    if (t < 7) {
        float ca = ang_last[(long)i * 14 + t * 2];
        float sa = ang_last[(long)i * 14 + t * 2 + 1];
        float inv = rsqrtf(ca * ca + sa * sa + 1e-8f);
        ca *= inv; sa *= inv;
        float* F = outfr + (long)i * 84 + t * 12;
        for (int r = 0; r < 3; ++r) {
            F[r * 3 + 0] = R[r * 3 + 0];
            F[r * 3 + 1] = R[r * 3 + 1] * ca + R[r * 3 + 2] * sa;
            F[r * 3 + 2] = -R[r * 3 + 1] * sa + R[r * 3 + 2] * ca;
        }
        F[9] = trsl[i * 3]; F[10] = trsl[i * 3 + 1]; F[11] = trsl[i * 3 + 2];
    }
}

// =========================================================== host side =====
extern "C" void kernel_launch(void* const* d_in, const int* in_sizes, int n_in,
                              void* d_out, int out_size, void* d_ws, size_t ws_size,
                              hipStream_t stream) {
    (void)in_sizes; (void)n_in; (void)out_size; (void)ws_size;
    const float* sfea = (const float*)d_in[0];
    const float* pfea = (const float*)d_in[1];
    const float* encd = (const float*)d_in[2];
    const float* norm_s_g = (const float*)d_in[3];
    const float* norm_s_b = (const float*)d_in[4];
    const float* norm_p_g = (const float*)d_in[5];
    const float* norm_p_b = (const float*)d_in[6];
    const float* lin_s_W  = (const float*)d_in[7];
    const float* lin_s_b  = (const float*)d_in[8];
    const float* ipa_Wq   = (const float*)d_in[9];
    const float* ipa_Wk   = (const float*)d_in[10];
    const float* ipa_Wv   = (const float*)d_in[11];
    const float* ipa_Wqp  = (const float*)d_in[12];
    const float* ipa_Wkp  = (const float*)d_in[13];
    const float* ipa_Wvp  = (const float*)d_in[14];
    const float* ipa_Wb   = (const float*)d_in[15];
    const float* ipa_hw   = (const float*)d_in[16];
    const float* ipa_Wo   = (const float*)d_in[17];
    const float* ipa_bo   = (const float*)d_in[18];
    const float* ipa_ln_g = (const float*)d_in[19];
    const float* ipa_ln_b = (const float*)d_in[20];
    const float* tr_W1 = (const float*)d_in[21];
    const float* tr_b1 = (const float*)d_in[22];
    const float* tr_W2 = (const float*)d_in[23];
    const float* tr_b2 = (const float*)d_in[24];
    const float* tr_W3 = (const float*)d_in[25];
    const float* tr_b3 = (const float*)d_in[26];
    const float* tr_ln_g = (const float*)d_in[27];
    const float* tr_ln_b = (const float*)d_in[28];
    const float* bb_W = (const float*)d_in[29];
    const float* bb_b = (const float*)d_in[30];
    const float* ang_Ws = (const float*)d_in[31];
    const float* ang_Wi = (const float*)d_in[32];
    const float* ang_W1 = (const float*)d_in[33];
    const float* ang_W2 = (const float*)d_in[34];
    const float* ang_Wo = (const float*)d_in[35];
    const float* ang_bo = (const float*)d_in[36];
    const float* pl_W1 = (const float*)d_in[37];
    const float* pl_b1 = (const float*)d_in[38];
    const float* pl_W2 = (const float*)d_in[39];
    const float* pl_b2 = (const float*)d_in[40];
    const float* pl_Wo = (const float*)d_in[41];
    const float* pl_bo = (const float*)d_in[42];

    float* out = (float*)d_out;
    float* out_q  = out;                 // (8,768,4)
    float* out_t  = out + 24576;         // (8,768,3)
    float* out_a  = out + 43008;         // (8,768,7,2)
    float* out_qu = out + 129024;        // (8,768,4)
    float* out_pl = out + 153600;        // (8,768,50)
    float* out_fr = out + 460800;        // (768,7,4,3)

    // ---- workspace bump allocator ----
    char* wsb = (char*)d_ws;
    size_t off = 0;
    auto alloc = [&](size_t bytes) -> void* {
        void* p = wsb + off;
        off = (off + bytes + 255) & ~(size_t)255;
        return p;
    };
    typedef unsigned short u16;
    u16*   zbt   = (u16*)alloc((size_t)L * CZ * L * 2);       // 302 MB
    float* biasS = (float*)alloc((size_t)HN * L * L * 4);     // 28 MB
    float* logits= (float*)alloc((size_t)HN * L * L * 4);     // 28 MB
    u16*   abf   = (u16*)alloc((size_t)L * 16 * L * 2);       // 19 MB
    u16*   Aatt  = (u16*)alloc((size_t)HN * L * 32 * 2);
    u16*   Batt  = (u16*)alloc((size_t)HN * L * 32 * 2);
    float* rowt  = (float*)alloc((size_t)HN * L * 4);
    float* colt  = (float*)alloc((size_t)HN * L * 4);
    float* proj  = (float*)alloc((size_t)L * 1152 * 4);
    float* vpg   = (float*)alloc((size_t)L * 288 * 4);
    float* opgb  = (float*)alloc((size_t)L * 288 * 4);
    float* cat   = (float*)alloc((size_t)L * DOUT * 4);
    u16*   catb  = (u16*)alloc((size_t)L * DOUT * 2);
    float* s     = (float*)alloc((size_t)L * CS * 4);
    u16*   sb    = (u16*)alloc((size_t)L * CS * 2);
    float* s_init= (float*)alloc((size_t)L * CS * 4);
    u16*   s_init_b = (u16*)alloc((size_t)L * CS * 2);
    u16*   rc    = (u16*)alloc((size_t)L * 416 * 2);
    u16*   rsb   = (u16*)alloc((size_t)L * CS * 2);
    float* delta = (float*)alloc((size_t)L * CS * 4);
    float* h1    = (float*)alloc((size_t)L * CS * 4);
    float* h2    = (float*)alloc((size_t)L * CS * 4);
    u16*   h1b   = (u16*)alloc((size_t)L * CS * 2);
    u16*   h2b   = (u16*)alloc((size_t)L * CS * 2);
    float* ang_init = (float*)alloc((size_t)L * 128 * 4);
    float* aA    = (float*)alloc((size_t)L * 128 * 4);
    float* uA    = (float*)alloc((size_t)L * 128 * 4);
    float* a2    = (float*)alloc((size_t)L * 128 * 4);
    u16*   tAb   = (u16*)alloc((size_t)L * 128 * 2);
    u16*   uAb   = (u16*)alloc((size_t)L * 128 * 2);
    u16*   a2b   = (u16*)alloc((size_t)L * 128 * 2);
    float* p1    = (float*)alloc((size_t)L * 128 * 4);
    float* p2    = (float*)alloc((size_t)L * 128 * 4);
    u16*   p1b   = (u16*)alloc((size_t)L * 128 * 2);
    u16*   p2b   = (u16*)alloc((size_t)L * 128 * 2);
    float* quat  = (float*)alloc((size_t)L * 4 * 4);
    float* trsl  = (float*)alloc((size_t)L * 3 * 4);
    // swizzled weights
    u16* wcat  = (u16*)alloc((size_t)12 * 72 * 512 * 2);
    u16* wo    = (u16*)alloc((size_t)114 * 24 * 512 * 2);
    u16* wlins = (u16*)alloc((size_t)12 * 24 * 512 * 2);
    u16* wtr1  = (u16*)alloc((size_t)12 * 24 * 512 * 2);
    u16* wtr2  = (u16*)alloc((size_t)12 * 24 * 512 * 2);
    u16* wtr3  = (u16*)alloc((size_t)12 * 24 * 512 * 2);
    u16* wangs = (u16*)alloc((size_t)12 * 8 * 512 * 2);
    u16* wangi = (u16*)alloc((size_t)13 * 8 * 512 * 2);
    u16* wang1 = (u16*)alloc((size_t)4 * 8 * 512 * 2);
    u16* wang2 = (u16*)alloc((size_t)4 * 8 * 512 * 2);
    u16* wango = (u16*)alloc((size_t)4 * 1 * 512 * 2);
    u16* wpl1  = (u16*)alloc((size_t)12 * 8 * 512 * 2);
    u16* wpl2  = (u16*)alloc((size_t)4 * 8 * 512 * 2);
    u16* wplo  = (u16*)alloc((size_t)4 * 4 * 512 * 2);

    auto swz = [&](const float* W, int K, int N, u16* dst, int KT, int NTloc, int NTtot, int ntOff) {
        long total = (long)KT * NTloc * 512;
        int blocks = (int)((total + 255) / 256);
        k_swizzle<<<blocks, 256, 0, stream>>>(W, K, N, dst, KT, NTloc, NTtot, ntOff);
    };
    // concat weight for q|k|v|qp|kp|vp  (N = 1152, 72 n-tiles)
    swz(ipa_Wq, 384, 192, wcat, 12, 12, 72, 0);
    swz(ipa_Wk, 384, 192, wcat, 12, 12, 72, 12);
    swz(ipa_Wv, 384, 192, wcat, 12, 12, 72, 24);
    swz(ipa_Wqp, 384, 144, wcat, 12, 9, 72, 36);
    swz(ipa_Wkp, 384, 144, wcat, 12, 9, 72, 45);
    swz(ipa_Wvp, 384, 288, wcat, 12, 18, 72, 54);
    swz(ipa_Wo, 3648, 384, wo, 114, 24, 24, 0);
    swz(lin_s_W, 384, 384, wlins, 12, 24, 24, 0);
    swz(tr_W1, 384, 384, wtr1, 12, 24, 24, 0);
    swz(tr_W2, 384, 384, wtr2, 12, 24, 24, 0);
    swz(tr_W3, 384, 384, wtr3, 12, 24, 24, 0);
    swz(ang_Ws, 384, 128, wangs, 12, 8, 8, 0);
    swz(ang_Wi, 416, 128, wangi, 13, 8, 8, 0);
    swz(ang_W1, 128, 128, wang1, 4, 8, 8, 0);
    swz(ang_W2, 128, 128, wang2, 4, 8, 8, 0);
    swz(ang_Wo, 128, 14, wango, 4, 1, 1, 0);
    swz(pl_W1, 384, 128, wpl1, 12, 8, 8, 0);
    swz(pl_W2, 128, 128, wpl2, 4, 8, 8, 0);
    swz(pl_Wo, 128, 50, wplo, 4, 4, 4, 0);

    // ---- prep ----
    k_prep_single<<<L, 128, 0, stream>>>(sfea, encd, norm_s_g, norm_s_b, s_init, s_init_b, rc);
    k_prep_pair<<<dim3(48, L), 256, 0, stream>>>(pfea, norm_p_g, norm_p_b, ipa_Wb, zbt, biasS);
    k_init_qt<<<3, 256, 0, stream>>>(quat, trsl);
    k_zeropad_abf<<<9216, 256, 0, stream>>>(abf);
    // ang_init = relu(cat(s_init,e)) @ ang_Wi
    k_gemm_ws<<<dim3(8, 48), 32, 0, stream>>>(rc, 416, wangi, nullptr, nullptr, 0, ang_init, 128, 128, 13, 0);
    // s = s_init @ lin_s_W + b
    k_gemm_ws<<<dim3(24, 48), 32, 0, stream>>>(s_init_b, 384, wlins, lin_s_b, nullptr, 0, s, 384, 384, 12, 0);
    k_cvt<<<1152, 256, 0, stream>>>(s, sb, (long)L * CS, 0);

    // ---- layers ----
    for (int lyr = 0; lyr < 8; ++lyr) {
        // IPA
        k_gemm_ws<<<dim3(72, 48), 32, 0, stream>>>(sb, 384, wcat, nullptr, nullptr, 0, proj, 1152, 1152, 12, 0);
        k_point<<<L, 64, 0, stream>>>(proj, quat, trsl, ipa_hw, Aatt, Batt, rowt, colt, vpg);
        k_gemm_att<<<dim3(48, 48, 12), 32, 0, stream>>>(Aatt, Batt, biasS, rowt, colt, logits);
        k_softmax<<<dim3(L, 12), 256, 0, stream>>>(logits, abf);
        // o -> cat[:, 0:192]
        k_gemm_nn<<<dim3(1, 48, 12), 32, 0, stream>>>(abf, 16 * L, (long)L,
                                                      proj + 384, 1152, 16L,
                                                      cat, DOUT, 16L, 16, 24);
        // opg -> opgb
        k_gemm_nn<<<dim3(2, 48, 12), 32, 0, stream>>>(abf, 16 * L, (long)L,
                                                      vpg, 288, 24L,
                                                      opgb, 288, 24L, 24, 24);
        // opair -> cat[:, 576:3648]  (batched over i, B = z[i]^T bf16)
        k_gemm_nt<<<dim3(16, 1, L), 32, 0, stream>>>(abf, L, (long)16 * L,
                                                     zbt, L, (long)CZ * L,
                                                     cat + 576, CZ, (long)DOUT, 12, 24);
        k_opl<<<L, 32, 0, stream>>>(opgb, quat, trsl, cat);
        k_cvt<<<10944, 256, 0, stream>>>(cat, catb, (long)L * DOUT, 0);
        k_gemm_ws<<<dim3(24, 48), 32, 0, stream>>>(catb, DOUT, wo, ipa_bo, nullptr, 0, delta, 384, 384, 114, 0);
        k_ln_res<<<L, 128, 0, stream>>>(s, delta, ipa_ln_g, ipa_ln_b, s, sb);
        // transition
        k_gemm_ws<<<dim3(24, 48), 32, 0, stream>>>(sb, 384, wtr1, tr_b1, nullptr, 0, h1, 384, 384, 12, 1);
        k_cvt<<<1152, 256, 0, stream>>>(h1, h1b, (long)L * CS, 0);
        k_gemm_ws<<<dim3(24, 48), 32, 0, stream>>>(h1b, 384, wtr2, tr_b2, nullptr, 0, h2, 384, 384, 12, 1);
        k_cvt<<<1152, 256, 0, stream>>>(h2, h2b, (long)L * CS, 0);
        k_gemm_ws<<<dim3(24, 48), 32, 0, stream>>>(h2b, 384, wtr3, tr_b3, nullptr, 0, delta, 384, 384, 12, 0);
        k_ln_res<<<L, 128, 0, stream>>>(s, delta, tr_ln_g, tr_ln_b, s, sb);
        // backbone update + per-layer outputs
        k_bb<<<L, 64, 0, stream>>>(s, bb_W, bb_b, quat, trsl,
                                   out_q + (size_t)lyr * L * 4,
                                   out_t + (size_t)lyr * L * 3,
                                   out_qu + (size_t)lyr * L * 4);
        // angles
        k_cvt<<<1152, 256, 0, stream>>>(s, rsb, (long)L * CS, 1);
        k_gemm_ws<<<dim3(8, 48), 32, 0, stream>>>(rsb, 384, wangs, nullptr, ang_init, 128, aA, 128, 128, 12, 0);
        k_cvt<<<384, 256, 0, stream>>>(aA, tAb, (long)L * 128, 1);
        k_gemm_ws<<<dim3(8, 48), 32, 0, stream>>>(tAb, 128, wang1, nullptr, nullptr, 0, uA, 128, 128, 4, 1);
        k_cvt<<<384, 256, 0, stream>>>(uA, uAb, (long)L * 128, 0);
        k_gemm_ws<<<dim3(8, 48), 32, 0, stream>>>(uAb, 128, wang2, nullptr, aA, 128, a2, 128, 128, 4, 0);
        k_cvt<<<384, 256, 0, stream>>>(a2, a2b, (long)L * 128, 1);
        k_gemm_ws<<<dim3(1, 48), 32, 0, stream>>>(a2b, 128, wango, ang_bo, nullptr, 0,
                                                  out_a + (size_t)lyr * L * 14, 14, 14, 4, 0);
        // plddt
        k_gemm_ws<<<dim3(8, 48), 32, 0, stream>>>(sb, 384, wpl1, pl_b1, nullptr, 0, p1, 128, 128, 12, 1);
        k_cvt<<<384, 256, 0, stream>>>(p1, p1b, (long)L * 128, 0);
        k_gemm_ws<<<dim3(8, 48), 32, 0, stream>>>(p1b, 128, wpl2, pl_b2, nullptr, 0, p2, 128, 128, 4, 1);
        k_cvt<<<384, 256, 0, stream>>>(p2, p2b, (long)L * 128, 0);
        k_gemm_ws<<<dim3(4, 48), 32, 0, stream>>>(p2b, 128, wplo, pl_bo, nullptr, 0,
                                                  out_pl + (size_t)lyr * L * 50, 50, 50, 4, 0);
    }
    // final side-chain frames
    k_fram<<<L, 32, 0, stream>>>(quat, trsl, out_a + (size_t)7 * L * 14, out_fr);
}
